// MixerBlock_59803124629578
// MI455X (gfx1250) — matrix operand order repaired and checked by reading, not compiled, not run on hardware
//
#include <hip/hip_runtime.h>
#include <hip/hip_bf16.h>

typedef __attribute__((ext_vector_type(16))) _Float16 v16h;
typedef __attribute__((ext_vector_type(8)))  float    v8f;

static constexpr int kB  = 4;
static constexpr int kN  = 10000;
static constexpr int kNE = 100000;
static constexpr int kD  = 128;
static constexpr int kT  = 128;
static constexpr int kH  = 4;

union FragU { v16h v; unsigned int u[8]; };

// A fragment: 16x32 f16, row m = lane%16, half h = lane/16
// VGPR p : K = (p<4) ? 8h+2p : 16+8h+2(p-4)   (pairs packed in one dword)
static __device__ inline v16h load_a_frag(const _Float16* At, int ld, int mbase, int l) {
  int m = mbase + (l & 15);
  int h = l >> 4;
  FragU f;
  const _Float16* row = At + m * ld;
#pragma unroll
  for (int p = 0; p < 8; ++p) {
    int k = (p < 4) ? (8 * h + 2 * p) : (16 + 8 * h + 2 * (p - 4));
    f.u[p] = *(const unsigned int*)(row + k);
  }
  return f.v;
}

// B fragment: 32x16 f16, col n = lane%16, half h = lane/16
// VGPR p : K = (p<4) ? 8h+2p : 16+8h+2(p-4). Bt stored [n][k] so pairs are contiguous dwords.
static __device__ inline v16h load_b_frag(const _Float16* Bt, int ld, int nbase, int l) {
  int n = nbase + (l & 15);
  int h = l >> 4;
  FragU f;
  const _Float16* row = Bt + n * ld;
#pragma unroll
  for (int p = 0; p < 8; ++p) f.u[p] = *(const unsigned int*)(row + ((p < 4) ? (8 * h + 2 * p) : (16 + 8 * h + 2 * (p - 4))));
  return f.v;
}

static __device__ inline v8f wmma16(v16h a, v16h b, v8f c) {
  return __builtin_amdgcn_wmma_f32_16x16x32_f16(false, a, false, b, (short)0, c, false, false);
}

static __device__ inline float silu(float x) { return x / (1.f + __expf(-x)); }

// vectorized staging helpers ------------------------------------------------
static __device__ inline void load16(float* buf, const float* __restrict__ src) {
  const float4* p = (const float4*)src;
  float4 a = p[0], b = p[1], c = p[2], d = p[3];
  buf[0]=a.x; buf[1]=a.y; buf[2]=a.z; buf[3]=a.w;
  buf[4]=b.x; buf[5]=b.y; buf[6]=b.z; buf[7]=b.w;
  buf[8]=c.x; buf[9]=c.y; buf[10]=c.z; buf[11]=c.w;
  buf[12]=d.x; buf[13]=d.y; buf[14]=d.z; buf[15]=d.w;
}
static __device__ inline void load8(float* buf, const float* __restrict__ src) {
  const float4* p = (const float4*)src;
  float4 a = p[0], b = p[1];
  buf[0]=a.x; buf[1]=a.y; buf[2]=a.z; buf[3]=a.w;
  buf[4]=b.x; buf[5]=b.y; buf[6]=b.z; buf[7]=b.w;
}
static __device__ inline void cvt_store16(_Float16* dst, const float* buf) {
  union { _Float16 h[16]; uint4 q[2]; } u;
#pragma unroll
  for (int j = 0; j < 16; ++j) u.h[j] = (_Float16)buf[j];
  ((uint4*)dst)[0] = u.q[0];
  ((uint4*)dst)[1] = u.q[1];
}
static __device__ inline void zero_store16(_Float16* dst) {
  uint4 z; z.x = z.y = z.z = z.w = 0u;
  ((uint4*)dst)[0] = z;
  ((uint4*)dst)[1] = z;
}

// ---------------------------------------------------------------------------
// Generic GEMM: out[m,n] = act(alpha * sum_k A[m,k]*B(k,n) + bias[n])
// bfmt==0 : Bw[n*ldb + k]   bfmt==1 : Bw[k*ldb + n]
// block = 128 threads, 64x64 tile, 4 waves of 16 rows each.
// Alignment contract: lda/ldb % 4 == 0, base pointers 16B aligned.
// ---------------------------------------------------------------------------
__global__ __launch_bounds__(128) void gemm_wmma(
    int M, int N, int K,
    const float* __restrict__ A, int lda,
    const float* __restrict__ Bw, int ldb, int bfmt,
    const float* __restrict__ bias,
    float* __restrict__ out, int ldo,
    float alpha, int act)
{
  __shared__ alignas(16) _Float16 At[64 * 40];
  __shared__ alignas(16) _Float16 Bt[64 * 40];
  const int tid = threadIdx.x;
  const int l = tid & 31, w = tid >> 5;
  const int rowBase = blockIdx.x * 64;
  const int colBase = blockIdx.y * 64;

  v8f acc[4];
#pragma unroll
  for (int t = 0; t < 4; ++t)
#pragma unroll
    for (int i = 0; i < 8; ++i) acc[t][i] = 0.f;

  const int ksteps = (K + 31) >> 5;
  for (int ks = 0; ks < ksteps; ++ks) {
    const int k0 = ks << 5;
    // ---- stage A: thread = (row i, 16-elem segment) ----
    {
      int i = tid >> 1, seg = tid & 1;
      int gm = rowBase + i;
      int gk0 = k0 + seg * 16;
      _Float16* dst = At + i * 40 + seg * 16;
      if (gm < M && gk0 + 16 <= K) {
        float buf[16];
        load16(buf, A + (long)gm * lda + gk0);
        cvt_store16(dst, buf);
      } else if (gm < M && gk0 < K) {
#pragma unroll
        for (int j = 0; j < 16; ++j) {
          int gk = gk0 + j;
          dst[j] = (_Float16)((gk < K) ? A[(long)gm * lda + gk] : 0.f);
        }
      } else {
        zero_store16(dst);
      }
    }
    // ---- stage B ----
    if (bfmt == 0) {
      int n = tid >> 1, seg = tid & 1;
      int gn = colBase + n;
      int gk0 = k0 + seg * 16;
      _Float16* dst = Bt + n * 40 + seg * 16;
      if (gn < N && gk0 + 16 <= K) {
        float buf[16];
        load16(buf, Bw + (long)gn * ldb + gk0);
        cvt_store16(dst, buf);
      } else if (gn < N && gk0 < K) {
#pragma unroll
        for (int j = 0; j < 16; ++j) {
          int gk = gk0 + j;
          dst[j] = (_Float16)((gk < K) ? Bw[(long)gn * ldb + gk] : 0.f);
        }
      } else {
        zero_store16(dst);
      }
    } else {
      // k-major B: thread = (k-pair, 8-col group); pack (k,k+1) per col.
      int kp = tid & 15;
      int ns = tid >> 4;
      int gk = k0 + 2 * kp;
      int gn0 = colBase + ns * 8;
      float r0[8], r1[8];
      if (gn0 + 8 <= N && gk + 2 <= K) {
        load8(r0, Bw + (long)gk * ldb + gn0);
        load8(r1, Bw + (long)(gk + 1) * ldb + gn0);
      } else {
#pragma unroll
        for (int j = 0; j < 8; ++j) {
          int gn = gn0 + j;
          r0[j] = (gn < N && gk < K) ? Bw[(long)gk * ldb + gn] : 0.f;
          r1[j] = (gn < N && gk + 1 < K) ? Bw[(long)(gk + 1) * ldb + gn] : 0.f;
        }
      }
#pragma unroll
      for (int j = 0; j < 8; ++j) {
        union { _Float16 h[2]; unsigned int u; } pk;
        pk.h[0] = (_Float16)r0[j];
        pk.h[1] = (_Float16)r1[j];
        *(unsigned int*)(Bt + (ns * 8 + j) * 40 + 2 * kp) = pk.u;
      }
    }
    __syncthreads();
    v16h af = load_a_frag(At, 40, w * 16, l);
#pragma unroll
    for (int t = 0; t < 4; ++t) {
      v16h bf = load_b_frag(Bt, 40, t * 16, l);
      acc[t] = wmma16(af, bf, acc[t]);
    }
    __syncthreads();
  }

  const int nloc = l & 15, h = l >> 4;
#pragma unroll
  for (int t = 0; t < 4; ++t) {
    int gn = colBase + t * 16 + nloc;
    if (gn >= N) continue;
    float bv = bias ? bias[gn] : 0.f;
#pragma unroll
    for (int p = 0; p < 8; ++p) {
      int gm = rowBase + w * 16 + h * 8 + p;
      if (gm < M) {
        float v = acc[t][p] * alpha + bv;
        if (act == 1) v = silu(v);
        out[(long)gm * ldo + gn] = v;
      }
    }
  }
}

// ---------------------------------------------------------------------------
// Fused edge MLP: gather(senders,receivers,E) -> L512x128 SiLU -> L128x128
// -> +bias -> LayerNorm -> E_out, scatter-add e0/e1 + counts.
// 64 edges / block, 4 waves x (16 rows x 128 cols).
// Gather regions [0,128)V[s] [128,192)senc[s] [192,320)V[r] [320,384)senc[r]
// [384,512)E -- boundaries are multiples of 64, so a 16-aligned 16-elem
// segment never crosses a region.
// ---------------------------------------------------------------------------
__global__ __launch_bounds__(128) void edge_mlp_kernel(
    const float* __restrict__ V, const float* __restrict__ E,
    const int* __restrict__ edges, const float* __restrict__ senc,
    const float* __restrict__ fw1, const float* __restrict__ fb1,
    const float* __restrict__ fw2, const float* __restrict__ fb2,
    const float* __restrict__ fg, const float* __restrict__ fbe,
    float* __restrict__ Eout,
    float* __restrict__ m0sum, float* __restrict__ m1sum,
    float* __restrict__ cnt0, float* __restrict__ cnt1)
{
  const int b = blockIdx.y;
  const int eBase = blockIdx.x * 64;
  const int tid = threadIdx.x;
  const int l = tid & 31, w = tid >> 5;

  __shared__ alignas(16) _Float16 At[64 * 40];
  __shared__ alignas(16) _Float16 Bt[128 * 40];
  __shared__ alignas(16) _Float16 Ht[64 * 136];
  __shared__ float Ot[64 * 129];
  __shared__ int sidx[64];
  __shared__ int ridx[64];
  __shared__ float validf[64];

  if (tid < 64) {
    int e = eBase + tid;
    int s = 0, r = 0; float va = 0.f;
    if (e < kNE) {
      long base = ((long)b * kNE + e) * 2;
      s = edges[base]; r = edges[base + 1];
      va = (s >= 0 && r >= 0) ? 1.f : 0.f;
      s = min(max(s, 0), kN - 1);
      r = min(max(r, 0), kN - 1);
    }
    sidx[tid] = s; ridx[tid] = r; validf[tid] = va;
  }
  __syncthreads();

  v8f acc[8];
#pragma unroll
  for (int t = 0; t < 8; ++t)
#pragma unroll
    for (int i = 0; i < 8; ++i) acc[t][i] = 0.f;

  // layer 1: K = 512
  for (int ks = 0; ks < 16; ++ks) {
    const int k0 = ks << 5;
    // stage A (gathered)
    {
      int i = tid >> 1, seg = tid & 1;
      int e = eBase + i;
      int kkb = k0 + seg * 16;
      _Float16* dst = At + i * 40 + seg * 16;
      if (e < kNE) {
        const float* src;
        if (kkb < 128)      src = V + ((long)b * kN + sidx[i]) * 128 + kkb;
        else if (kkb < 192) src = senc + ((long)b * kN + sidx[i]) * 64 + (kkb - 128);
        else if (kkb < 320) src = V + ((long)b * kN + ridx[i]) * 128 + (kkb - 192);
        else if (kkb < 384) src = senc + ((long)b * kN + ridx[i]) * 64 + (kkb - 320);
        else                src = E + ((long)b * kNE + e) * 128 + (kkb - 384);
        float buf[16];
        load16(buf, src);
        cvt_store16(dst, buf);
      } else {
        zero_store16(dst);
      }
    }
    // stage B: one weight row per thread (fw1: 128 x 512)
    {
      const float* src = fw1 + tid * 512 + k0;
      float buf[16];
      load16(buf, src);
      cvt_store16(Bt + tid * 40, buf);
      load16(buf, src + 16);
      cvt_store16(Bt + tid * 40 + 16, buf);
    }
    __syncthreads();
    v16h af = load_a_frag(At, 40, w * 16, l);
#pragma unroll
    for (int t = 0; t < 8; ++t) {
      v16h bf = load_b_frag(Bt, 40, t * 16, l);
      acc[t] = wmma16(af, bf, acc[t]);
    }
    __syncthreads();
  }

  // SiLU -> Ht (f16, row-major, ld 136)
  {
    int nloc = l & 15, h = l >> 4;
#pragma unroll
    for (int t = 0; t < 8; ++t) {
      int n = t * 16 + nloc;
      float bv = fb1[n];
#pragma unroll
      for (int p = 0; p < 8; ++p) {
        int m = w * 16 + h * 8 + p;
        Ht[m * 136 + n] = (_Float16)silu(acc[t][p] + bv);
      }
    }
  }
  __syncthreads();

  // layer 2: K = 128
  v8f a2[8];
#pragma unroll
  for (int t = 0; t < 8; ++t)
#pragma unroll
    for (int i = 0; i < 8; ++i) a2[t][i] = 0.f;

  for (int ks = 0; ks < 4; ++ks) {
    const int k0 = ks << 5;
    {
      const float* src = fw2 + tid * 128 + k0;
      float buf[16];
      load16(buf, src);
      cvt_store16(Bt + tid * 40, buf);
      load16(buf, src + 16);
      cvt_store16(Bt + tid * 40 + 16, buf);
    }
    __syncthreads();
    v16h af = load_a_frag(Ht + k0, 136, w * 16, l);
#pragma unroll
    for (int t = 0; t < 8; ++t) {
      v16h bf = load_b_frag(Bt, 40, t * 16, l);
      a2[t] = wmma16(af, bf, a2[t]);
    }
    __syncthreads();
  }

  {
    int nloc = l & 15, h = l >> 4;
#pragma unroll
    for (int t = 0; t < 8; ++t) {
      int n = t * 16 + nloc;
      float bv = fb2[n];
#pragma unroll
      for (int p = 0; p < 8; ++p) {
        int m = w * 16 + h * 8 + p;
        Ot[m * 129 + n] = a2[t][p] + bv;
      }
    }
  }
  __syncthreads();

  // LayerNorm + E_out + scatter-add (2 threads per row, 64 features each)
  {
    int row = tid >> 1, half = tid & 1;
    int e = eBase + row;
    if (e < kNE) {
      const float* orow = Ot + row * 129;
      float sum = 0.f;
      for (int f = 0; f < 128; ++f) sum += orow[f];
      float mean = sum * (1.f / 128.f);
      float vs = 0.f;
      for (int f = 0; f < 128; ++f) { float d = orow[f] - mean; vs += d * d; }
      float inv = rsqrtf(vs * (1.f / 128.f) + 1e-5f);
      int s = sidx[row], r = ridx[row];
      float va = validf[row];
      long ebase = ((long)b * kNE + e) * 128;
      for (int f = half * 64; f < half * 64 + 64; ++f) {
        float em = ((orow[f] - mean) * inv * fg[f] + fbe[f]) * va;
        Eout[ebase + f] = E[ebase + f] + em;
        if (f < 64) atomicAdd(&m0sum[((long)b * kN + s) * 64 + f], em);
        else        atomicAdd(&m1sum[((long)b * kN + r) * 64 + (f - 64)], em);
      }
      if (half == 0) atomicAdd(&cnt0[b * kN + s], 1.f);
      else           atomicAdd(&cnt1[b * kN + r], 1.f);
    }
  }
}

// ---------------------------------------------------------------------------
// Fused node MLP: (V|senc|m0|m1) -> L320x128 SiLU -> L128x128 -> LN -> v_emb
// V1 = V + v_emb ; W0 = LN(V1, ln1)
// ---------------------------------------------------------------------------
__global__ __launch_bounds__(128) void node_mlp_kernel(
    const float* __restrict__ V, const float* __restrict__ senc,
    const float* __restrict__ m0sum, const float* __restrict__ m1sum,
    const float* __restrict__ cnt0, const float* __restrict__ cnt1,
    const float* __restrict__ fw1, const float* __restrict__ fb1,
    const float* __restrict__ fw2, const float* __restrict__ fb2,
    const float* __restrict__ fg, const float* __restrict__ fbe,
    const float* __restrict__ ln1g, const float* __restrict__ ln1b,
    float* __restrict__ V1g, float* __restrict__ W0g)
{
  const int b = blockIdx.y;
  const int nBase = blockIdx.x * 64;
  const int tid = threadIdx.x;
  const int l = tid & 31, w = tid >> 5;

  __shared__ alignas(16) _Float16 At[64 * 40];
  __shared__ alignas(16) _Float16 Bt[128 * 40];
  __shared__ alignas(16) _Float16 Ht[64 * 136];
  __shared__ float Ot[64 * 129];

  v8f acc[8];
#pragma unroll
  for (int t = 0; t < 8; ++t)
#pragma unroll
    for (int i = 0; i < 8; ++i) acc[t][i] = 0.f;

  // layer 1: K = 320 (regions at 128/192/256 are multiples of 64)
  for (int ks = 0; ks < 10; ++ks) {
    const int k0 = ks << 5;
    {
      int i = tid >> 1, seg = tid & 1;
      int node = nBase + i;
      int kkb = k0 + seg * 16;
      _Float16* dst = At + i * 40 + seg * 16;
      if (node < kN) {
        long nb = (long)b * kN + node;
        const float* src;
        float scale = 1.f;
        if (kkb < 128)      src = V + nb * 128 + kkb;
        else if (kkb < 192) src = senc + nb * 64 + (kkb - 128);
        else if (kkb < 256) { src = m0sum + nb * 64 + (kkb - 192); scale = 1.f / fmaxf(cnt0[nb], 1.f); }
        else                { src = m1sum + nb * 64 + (kkb - 256); scale = 1.f / fmaxf(cnt1[nb], 1.f); }
        float buf[16];
        load16(buf, src);
#pragma unroll
        for (int j = 0; j < 16; ++j) buf[j] *= scale;
        cvt_store16(dst, buf);
      } else {
        zero_store16(dst);
      }
    }
    {
      const float* src = fw1 + tid * 320 + k0;
      float buf[16];
      load16(buf, src);
      cvt_store16(Bt + tid * 40, buf);
      load16(buf, src + 16);
      cvt_store16(Bt + tid * 40 + 16, buf);
    }
    __syncthreads();
    v16h af = load_a_frag(At, 40, w * 16, l);
#pragma unroll
    for (int t = 0; t < 8; ++t) {
      v16h bf = load_b_frag(Bt, 40, t * 16, l);
      acc[t] = wmma16(af, bf, acc[t]);
    }
    __syncthreads();
  }

  {
    int nloc = l & 15, h = l >> 4;
#pragma unroll
    for (int t = 0; t < 8; ++t) {
      int n = t * 16 + nloc;
      float bv = fb1[n];
#pragma unroll
      for (int p = 0; p < 8; ++p) {
        int m = w * 16 + h * 8 + p;
        Ht[m * 136 + n] = (_Float16)silu(acc[t][p] + bv);
      }
    }
  }
  __syncthreads();

  v8f a2[8];
#pragma unroll
  for (int t = 0; t < 8; ++t)
#pragma unroll
    for (int i = 0; i < 8; ++i) a2[t][i] = 0.f;

  for (int ks = 0; ks < 4; ++ks) {
    const int k0 = ks << 5;
    {
      const float* src = fw2 + tid * 128 + k0;
      float buf[16];
      load16(buf, src);
      cvt_store16(Bt + tid * 40, buf);
      load16(buf, src + 16);
      cvt_store16(Bt + tid * 40 + 16, buf);
    }
    __syncthreads();
    v16h af = load_a_frag(Ht + k0, 136, w * 16, l);
#pragma unroll
    for (int t = 0; t < 8; ++t) {
      v16h bf = load_b_frag(Bt, 40, t * 16, l);
      a2[t] = wmma16(af, bf, a2[t]);
    }
    __syncthreads();
  }

  {
    int nloc = l & 15, h = l >> 4;
#pragma unroll
    for (int t = 0; t < 8; ++t) {
      int n = t * 16 + nloc;
      float bv = fb2[n];
#pragma unroll
      for (int p = 0; p < 8; ++p) {
        int m = w * 16 + h * 8 + p;
        Ot[m * 129 + n] = a2[t][p] + bv;
      }
    }
  }
  __syncthreads();

  // LN -> v_emb ; V1 = V + v_emb ; W0 = LN(V1)
  if (tid < 64) {
    int row = tid;
    int node = nBase + row;
    if (node < kN) {
      float* orow = Ot + row * 129;
      long nb = (long)b * kN + node;
      float sum = 0.f;
      for (int f = 0; f < 128; ++f) sum += orow[f];
      float mean = sum * (1.f / 128.f);
      float vs = 0.f;
      for (int f = 0; f < 128; ++f) { float d = orow[f] - mean; vs += d * d; }
      float inv = rsqrtf(vs * (1.f / 128.f) + 1e-5f);
      for (int f = 0; f < 128; ++f) {
        float ve = (orow[f] - mean) * inv * fg[f] + fbe[f];
        float v1 = V[nb * 128 + f] + ve;
        orow[f] = v1;
        V1g[nb * 128 + f] = v1;
      }
      float s2 = 0.f;
      for (int f = 0; f < 128; ++f) s2 += orow[f];
      float m2 = s2 * (1.f / 128.f);
      float vs2 = 0.f;
      for (int f = 0; f < 128; ++f) { float d = orow[f] - m2; vs2 += d * d; }
      float inv2 = rsqrtf(vs2 * (1.f / 128.f) + 1e-5f);
      for (int f = 0; f < 128; ++f)
        W0g[nb * 128 + f] = (orow[f] - m2) * inv2 * ln1g[f] + ln1b[f];
    }
  }
}

// ---------------------------------------------------------------------------
// elementwise helpers
// ---------------------------------------------------------------------------
__global__ void zero_kernel(float* __restrict__ p, long n) {
  long i = (long)blockIdx.x * blockDim.x + threadIdx.x;
  long stride = (long)gridDim.x * blockDim.x;
  for (; i < n; i += stride) p[i] = 0.f;
}

__global__ void add_kernel(const float* __restrict__ a, const float* __restrict__ b,
                           float* __restrict__ c, long n) {
  long i = (long)blockIdx.x * blockDim.x + threadIdx.x;
  long stride = (long)gridDim.x * blockDim.x;
  for (; i < n; i += stride) c[i] = a[i] + b[i];
}

__global__ __launch_bounds__(128) void ln_kernel(const float* __restrict__ x,
                                                 const float* __restrict__ g,
                                                 const float* __restrict__ bcoef,
                                                 float* __restrict__ out) {
  int row = blockIdx.x;
  const float* xr = x + (long)row * 128;
  __shared__ float red[128];
  int t = threadIdx.x;
  float v = xr[t];
  red[t] = v;
  __syncthreads();
  for (int s = 64; s > 0; s >>= 1) { if (t < s) red[t] += red[t + s]; __syncthreads(); }
  float mean = red[0] * (1.f / 128.f);
  __syncthreads();
  float d = v - mean;
  red[t] = d * d;
  __syncthreads();
  for (int s = 64; s > 0; s >>= 1) { if (t < s) red[t] += red[t + s]; __syncthreads(); }
  float inv = rsqrtf(red[0] * (1.f / 128.f) + 1e-5f);
  out[(long)row * 128 + t] = d * inv * g[t] + bcoef[t];
}

__global__ __launch_bounds__(256) void softmax_kernel(float* __restrict__ S, int L) {
  long row = blockIdx.x;
  float* p = S + row * (long)L;
  __shared__ float red[256];
  int t = threadIdx.x;
  float mx = -3.0e38f;
  for (int i = t; i < L; i += 256) mx = fmaxf(mx, p[i]);
  red[t] = mx;
  __syncthreads();
  for (int s = 128; s > 0; s >>= 1) { if (t < s) red[t] = fmaxf(red[t], red[t + s]); __syncthreads(); }
  mx = red[0];
  __syncthreads();
  float sum = 0.f;
  for (int i = t; i < L; i += 256) { float e = __expf(p[i] - mx); p[i] = e; sum += e; }
  red[t] = sum;
  __syncthreads();
  for (int s = 128; s > 0; s >>= 1) { if (t < s) red[t] += red[t + s]; __syncthreads(); }
  float inv = 1.f / red[0];
  for (int i = t; i < L; i += 256) p[i] *= inv;
}

// ---------------------------------------------------------------------------
extern "C" void kernel_launch(void* const* d_in, const int* in_sizes, int n_in,
                              void* d_out, int out_size, void* d_ws, size_t ws_size,
                              hipStream_t stream) {
  (void)in_sizes; (void)n_in; (void)out_size; (void)ws_size;

  const float* V      = (const float*)d_in[0];
  const float* E      = (const float*)d_in[1];
  const int*   edges  = (const int*)  d_in[2];
  const float* senc   = (const float*)d_in[3];
  const float* fe_w1  = (const float*)d_in[4];
  const float* fe_b1  = (const float*)d_in[5];
  const float* fe_w2  = (const float*)d_in[6];
  const float* fe_b2  = (const float*)d_in[7];
  const float* fe_g   = (const float*)d_in[8];
  const float* fe_be  = (const float*)d_in[9];
  const float* fn_w1  = (const float*)d_in[10];
  const float* fn_b1  = (const float*)d_in[11];
  const float* fn_w2  = (const float*)d_in[12];
  const float* fn_b2  = (const float*)d_in[13];
  const float* fn_g   = (const float*)d_in[14];
  const float* fn_be  = (const float*)d_in[15];
  const float* ln1_g  = (const float*)d_in[16];
  const float* ln1_b  = (const float*)d_in[17];
  const float* ln2_g  = (const float*)d_in[18];
  const float* ln2_b  = (const float*)d_in[19];
  const float* Q      = (const float*)d_in[20];
  const float* a1_wqkv = (const float*)d_in[21];
  const float* a1_bqkv = (const float*)d_in[22];
  const float* a1_wo   = (const float*)d_in[23];
  const float* a1_bo   = (const float*)d_in[24];
  const float* a2_wqkv = (const float*)d_in[25];
  const float* a2_bqkv = (const float*)d_in[26];
  const float* a2_wo   = (const float*)d_in[27];
  const float* a2_bo   = (const float*)d_in[28];
  const float* a3_wqkv = (const float*)d_in[29];
  const float* a3_bqkv = (const float*)d_in[30];
  const float* a3_wo   = (const float*)d_in[31];
  const float* a3_bo   = (const float*)d_in[32];
  const float* ffn_w1  = (const float*)d_in[33];
  const float* ffn_b1  = (const float*)d_in[34];
  const float* ffn_w2  = (const float*)d_in[35];
  const float* ffn_b2  = (const float*)d_in[36];

  float* outp = (float*)d_out;
  float* V3out = outp;                               // B*N*128
  float* Eout  = outp + (size_t)kB * kN * 128;       // B*NE*128

  // workspace layout
  float* ws = (float*)d_ws;
  size_t off = 0;
  auto alloc = [&](size_t n) { float* p = ws + off; off += n; return p; };
  const size_t BND = (size_t)kB * kN * 128;
  float* m0sum = alloc((size_t)kB * kN * 64);
  float* m1sum = alloc((size_t)kB * kN * 64);
  float* cnt0  = alloc((size_t)kB * kN);
  float* cnt1  = alloc((size_t)kB * kN);
  const size_t zeroN = (size_t)kB * kN * 64 * 2 + (size_t)kB * kN * 2;
  float* wsV1  = alloc(BND);
  float* wsW0  = alloc(BND);
  float* qh1   = alloc((size_t)kT * kD);
  float* Kp    = alloc(BND);
  float* Vp    = alloc(BND);
  float* S1    = alloc((size_t)kB * kH * kT * kN);   // reused as S3
  float* O1    = alloc((size_t)kB * kT * kD);
  float* W1    = alloc((size_t)kB * kT * kD);
  float* QKV2  = alloc((size_t)kB * kT * 3 * kD);
  float* S2    = alloc((size_t)kB * kH * kT * kT);
  float* O2    = alloc((size_t)kB * kT * kD);
  float* W2    = alloc((size_t)kB * kT * kD);
  float* qh3   = alloc(BND);
  float* kh3   = alloc((size_t)kB * kT * kD);
  float* vh3   = alloc((size_t)kB * kT * kD);
  float* O3    = alloc(BND);
  float* AO    = alloc(BND);
  float* wsV2  = alloc(BND);
  float* F1    = alloc((size_t)kB * kN * 256);
  float* F2    = alloc(BND);
  float* S3    = S1;                                 // alias (S1 dead before S3 written)
  float* wsX   = Kp;                                 // alias (Kp dead before X written)

  const float iscale = 0.17677669529663687f;         // 1/sqrt(32)

  auto gemm = [&](int M, int N, int K, const float* A, int lda,
                  const float* Bw, int ldb, int bfmt, const float* bias,
                  float* out, int ldo, float alpha, int act) {
    dim3 g((M + 63) / 64, (N + 63) / 64, 1);
    gemm_wmma<<<g, 128, 0, stream>>>(M, N, K, A, lda, Bw, ldb, bfmt, bias, out, ldo, alpha, act);
  };

  // ---- GNN ----
  zero_kernel<<<1024, 256, 0, stream>>>(m0sum, (long)zeroN);
  {
    dim3 g((kNE + 63) / 64, kB, 1);
    edge_mlp_kernel<<<g, 128, 0, stream>>>(V, E, edges, senc, fe_w1, fe_b1, fe_w2, fe_b2,
                                           fe_g, fe_be, Eout, m0sum, m1sum, cnt0, cnt1);
  }
  {
    dim3 g((kN + 63) / 64, kB, 1);
    node_mlp_kernel<<<g, 128, 0, stream>>>(V, senc, m0sum, m1sum, cnt0, cnt1,
                                           fn_w1, fn_b1, fn_w2, fn_b2, fn_g, fn_be,
                                           ln1_g, ln1_b, wsV1, wsW0);
  }

  // ---- MHA 1 : q = Q (broadcast), k=v=W0 ----
  gemm(kT, kD, kD, Q, kD, a1_wqkv, kD, 0, a1_bqkv, qh1, kD, 1.f, 0);
  gemm(kB * kN, kD, kD, wsW0, kD, a1_wqkv + kD * kD, kD, 0, a1_bqkv + kD, Kp, kD, 1.f, 0);
  gemm(kB * kN, kD, kD, wsW0, kD, a1_wqkv + 2 * kD * kD, kD, 0, a1_bqkv + 2 * kD, Vp, kD, 1.f, 0);
  for (int b = 0; b < kB; ++b)
    for (int h = 0; h < kH; ++h)
      gemm(kT, kN, 32, qh1 + h * 32, kD,
           Kp + (size_t)b * kN * kD + h * 32, kD, 0, nullptr,
           S1 + ((size_t)(b * kH + h)) * kT * kN, kN, iscale, 0);
  softmax_kernel<<<kB * kH * kT, 256, 0, stream>>>(S1, kN);
  for (int b = 0; b < kB; ++b)
    for (int h = 0; h < kH; ++h)
      gemm(kT, 32, kN, S1 + ((size_t)(b * kH + h)) * kT * kN, kN,
           Vp + (size_t)b * kN * kD + h * 32, kD, 1, nullptr,
           O1 + (size_t)b * kT * kD + h * 32, kD, 1.f, 0);
  gemm(kB * kT, kD, kD, O1, kD, a1_wo, kD, 0, a1_bo, W1, kD, 1.f, 0);

  // ---- MHA 2 : self-attention on W1 (B x 128 x 128) ----
  gemm(kB * kT, 3 * kD, kD, W1, kD, a2_wqkv, kD, 0, a2_bqkv, QKV2, 3 * kD, 1.f, 0);
  for (int b = 0; b < kB; ++b)
    for (int h = 0; h < kH; ++h)
      gemm(kT, kT, 32, QKV2 + (size_t)b * kT * 384 + h * 32, 384,
           QKV2 + (size_t)b * kT * 384 + kD + h * 32, 384, 0, nullptr,
           S2 + ((size_t)(b * kH + h)) * kT * kT, kT, iscale, 0);
  softmax_kernel<<<kB * kH * kT, 256, 0, stream>>>(S2, kT);
  for (int b = 0; b < kB; ++b)
    for (int h = 0; h < kH; ++h)
      gemm(kT, 32, kT, S2 + ((size_t)(b * kH + h)) * kT * kT, kT,
           QKV2 + (size_t)b * kT * 384 + 2 * kD + h * 32, 384, 1, nullptr,
           O2 + (size_t)b * kT * kD + h * 32, kD, 1.f, 0);
  gemm(kB * kT, kD, kD, O2, kD, a2_wo, kD, 0, a2_bo, W2, kD, 1.f, 0);

  // ---- MHA 3 : q = W0 (B x N), k=v=W2 ----
  gemm(kB * kN, kD, kD, wsW0, kD, a3_wqkv, kD, 0, a3_bqkv, qh3, kD, 1.f, 0);
  gemm(kB * kT, kD, kD, W2, kD, a3_wqkv + kD * kD, kD, 0, a3_bqkv + kD, kh3, kD, 1.f, 0);
  gemm(kB * kT, kD, kD, W2, kD, a3_wqkv + 2 * kD * kD, kD, 0, a3_bqkv + 2 * kD, vh3, kD, 1.f, 0);
  for (int b = 0; b < kB; ++b)
    for (int h = 0; h < kH; ++h)
      gemm(kN, kT, 32, qh3 + (size_t)b * kN * kD + h * 32, kD,
           kh3 + (size_t)b * kT * kD + h * 32, kD, 0, nullptr,
           S3 + ((size_t)(b * kH + h)) * kN * kT, kT, iscale, 0);
  softmax_kernel<<<kB * kH * kN, 256, 0, stream>>>(S3, kT);
  for (int b = 0; b < kB; ++b)
    for (int h = 0; h < kH; ++h)
      gemm(kN, 32, kT, S3 + ((size_t)(b * kH + h)) * kN * kT, kT,
           vh3 + (size_t)b * kT * kD + h * 32, kD, 1, nullptr,
           O3 + (size_t)b * kN * kD + h * 32, kD, 1.f, 0);
  gemm(kB * kN, kD, kD, O3, kD, a3_wo, kD, 0, a3_bo, AO, kD, 1.f, 0);

  // ---- residual + FFN ----
  add_kernel<<<2048, 256, 0, stream>>>(wsV1, AO, wsV2, (long)BND);
  ln_kernel<<<kB * kN, 128, 0, stream>>>(wsV2, ln2_g, ln2_b, wsX);
  gemm(kB * kN, 2 * kD, kD, wsX, kD, ffn_w1, kD, 0, ffn_b1, F1, 2 * kD, 1.f, 1);
  gemm(kB * kN, kD, 2 * kD, F1, 2 * kD, ffn_w2, 2 * kD, 0, ffn_b2, F2, kD, 1.f, 0);
  add_kernel<<<2048, 256, 0, stream>>>(wsV2, F2, V3out, (long)BND);
}
